// NAFBlock1D_22608707846331
// MI455X (gfx1250) — matrix operand order repaired and checked by reading, not compiled, not run on hardware
//
#include <hip/hip_runtime.h>

// ---------------- types ----------------
typedef _Float16 half_t;
typedef __attribute__((ext_vector_type(16))) _Float16 v16h;
typedef __attribute__((ext_vector_type(8)))  _Float16 v8h;
typedef __attribute__((ext_vector_type(8)))  float    v8f;
typedef __attribute__((ext_vector_type(4)))  int      v4i;

// ---------------- problem constants ----------------
#define BB   16
#define CC   256
#define CC2  512
#define LL   8192
#define EPSV 1e-5f

// K1 tiling: 128 columns (8 WMMA n-tiles), interior 126, halo 1 each side
#define NT1       128
#define INT1      126
#define N1TILES   66          // ceil(8192/126)
#define YN_STRIDE 264         // f16 elems/row: 16B-aligned chunks, bank-conflict-free
#define PW_STRIDE 130

// K4 tiling
#define NT2       128
#define A_STRIDE  264
#define GT_STRIDE 264

// ---------------- optional CDNA5 async-to-LDS path (probe-guarded) ----------------
#if defined(__HIP_DEVICE_COMPILE__) &&                                           \
    __has_builtin(__builtin_amdgcn_global_load_async_to_lds_b128) &&             \
    __has_builtin(__builtin_amdgcn_s_wait_asynccnt)
#define USE_ASYNC_LDS 1
typedef __attribute__((address_space(1))) v4i* gas_v4i_t;   // global src, v4i, non-const
typedef __attribute__((address_space(3))) v4i* las_v4i_t;   // LDS dst, v4i
#endif

// ---------------- WMMA fragment helpers ----------------
// A (16x32 f16): lane = M row (lane%16); half h=lane/16 selects K:
//   elements 0..7  = K (8h .. 8h+7), elements 8..15 = K (16+8h .. 16+8h+7)
__device__ __forceinline__ v16h load_frag_a(const half_t* row_k0, int h) {
    v16h r;
    *((v8h*)&r)     = *(const v8h*)(row_k0 + 8 * h);
    *((v8h*)&r + 1) = *(const v8h*)(row_k0 + 16 + 8 * h);
    return r;
}
// B (32x16 f16): lane = N col (lane%16); half h: elements 0..7 = K (8h..8h+7), 8..15 = K (16+8h..16+8h+7)
__device__ __forceinline__ v16h load_frag_b(const half_t* col_k0, int h) {
    v16h r;
    *((v8h*)&r)     = *(const v8h*)(col_k0 + 8 * h);
    *((v8h*)&r + 1) = *(const v8h*)(col_k0 + 16 + 8 * h);
    return r;
}
__device__ __forceinline__ v8f wmma_f16(v16h a, v16h b, v8f c) {
    return __builtin_amdgcn_wmma_f32_16x16x32_f16(false, a, false, b, (short)0, c,
                                                  false, false);
}
__device__ __forceinline__ float sigmoidf_(float x) {
    return 1.0f / (1.0f + __expf(-x));
}

// ---------------- K0: w1 fp32 -> f16 ----------------
__global__ __launch_bounds__(256) void k_convert_w1(const float* __restrict__ w1,
                                                    half_t* __restrict__ w1h) {
    int idx = blockIdx.x * 256 + threadIdx.x;
    if (idx < CC2 * CC) w1h[idx] = (half_t)w1[idx];
}

// ---------------- K1: LayerNorm + pw1(WMMA) + depthwise + gate ----------------
// grid = (66, 16), block = 512 (16 waves)
__global__ __launch_bounds__(512) void k_ln_pw1_dw_gate(
    const float* __restrict__ x, const float* __restrict__ gamma,
    const float* __restrict__ beta, const half_t* __restrict__ w1h,
    const float* __restrict__ b1, const float* __restrict__ wd,
    const float* __restrict__ bd, half_t* __restrict__ g) {
    extern __shared__ char smem[];
    half_t* ynT  = (half_t*)smem;                              // [128][264] (n, k)
    half_t* pw1  = (half_t*)(smem + NT1 * YN_STRIDE * 2);      // [512][130]
    float*  sums = (float*)(smem + NT1 * YN_STRIDE * 2 + CC2 * PW_STRIDE * 2);
    float*  sqs  = sums + 4 * NT1;
    float*  mu   = sqs + 4 * NT1;
    float*  rstd = mu + NT1;

    const int tid  = threadIdx.x;
    const int tile = blockIdx.x;
    const int b    = blockIdx.y;
    const int l0   = tile * INT1 - 1;  // l of column 0 (halo)

    // ---- stage x (f16, transposed) + LN statistics ----
    const int col = tid & (NT1 - 1);   // 0..127
    const int seg = tid >> 7;          // 0..3, 64 rows each
    const int l   = l0 + col;
    const bool lok = (l >= 0) && (l < LL);
    float s = 0.f, sq = 0.f;
    if (lok) {
        const float* xp = x + ((size_t)b * CC + seg * 64) * LL + l;
        for (int i = 0; i < 64; ++i) {
            float v = xp[(size_t)i * LL];
            s += v; sq += v * v;
            ynT[col * YN_STRIDE + seg * 64 + i] = (half_t)v;
        }
    } else {
        for (int i = 0; i < 64; ++i) ynT[col * YN_STRIDE + seg * 64 + i] = (half_t)0.f;
    }
    sums[seg * NT1 + col] = s;
    sqs[seg * NT1 + col]  = sq;
    __syncthreads();
    if (tid < NT1) {
        float ts = 0.f, tq = 0.f;
        for (int sg = 0; sg < 4; ++sg) { ts += sums[sg * NT1 + tid]; tq += sqs[sg * NT1 + tid]; }
        float m = ts * (1.0f / CC);
        float v = tq * (1.0f / CC) - m * m;
        mu[tid]   = m;
        rstd[tid] = rsqrtf(v + EPSV);
    }
    __syncthreads();
    // ---- normalize in place (affine) ----
    {
        float m = mu[col], rs = rstd[col];
        for (int i = 0; i < 64; ++i) {
            int r = seg * 64 + i;
            float v = (float)ynT[col * YN_STRIDE + r];
            ynT[col * YN_STRIDE + r] = (half_t)((v - m) * rs * gamma[r] + beta[r]);
        }
    }
    __syncthreads();

    // ---- pw1 GEMM: M=512, K=256, N=128; m-tile outer, A-frag reused over 8 n-tiles ----
    const int wave = tid >> 5;         // 0..15
    const int lane = tid & 31;
    const int ln   = lane & 15;
    const int lh   = lane >> 4;
    for (int mt = wave; mt < 32; mt += 16) {   // 2 m-tiles per wave
        const int m0 = mt * 16;
        v8f acc[8] = {};
        const half_t* arow = w1h + (size_t)(m0 + ln) * CC;   // row M, K-major
        __builtin_prefetch(arow, 0, 0);
#pragma unroll
        for (int k0 = 0; k0 < CC; k0 += 32) {
            v16h a = load_frag_a(arow + k0, lh);
#pragma unroll
            for (int nt = 0; nt < 8; ++nt) {
                v16h bfr = load_frag_b(ynT + (nt * 16 + ln) * YN_STRIDE + k0, lh);
                acc[nt] = wmma_f16(a, bfr, acc[nt]);
            }
        }
        float bias[8];
#pragma unroll
        for (int v = 0; v < 8; ++v) bias[v] = b1[m0 + v + 8 * lh];
#pragma unroll
        for (int nt = 0; nt < 8; ++nt)
#pragma unroll
            for (int v = 0; v < 8; ++v) {
                int row = m0 + v + 8 * lh;
                pw1[row * PW_STRIDE + nt * 16 + ln] = (half_t)(acc[nt][v] + bias[v]);
            }
    }
    __syncthreads();

    // ---- depthwise conv (k=3, zero pad at L edges) + SimpleGate -> g ----
    for (int idx = tid; idx < CC * INT1; idx += 512) {
        const int ch = idx / INT1;
        const int c  = idx % INT1 + 1;         // interior column in tile
        const int lg = l0 + c;
        if (lg >= LL) continue;
        float z1 = bd[ch], z2 = bd[ch + CC];
#pragma unroll
        for (int j = 0; j < 3; ++j) {
            const int lc = c - 1 + j;
            const int la = l0 + lc;
            const bool ok = (la >= 0) && (la < LL);
            float v1 = ok ? (float)pw1[ch * PW_STRIDE + lc] : 0.f;
            float v2 = ok ? (float)pw1[(ch + CC) * PW_STRIDE + lc] : 0.f;
            z1 += wd[ch * 3 + j] * v1;
            z2 += wd[(ch + CC) * 3 + j] * v2;
        }
        g[((size_t)b * CC + ch) * LL + lg] = (half_t)(z1 * z2);
    }
}

// ---------------- K2: pool g over L ----------------
// grid = (256, 16), block = 256
__global__ __launch_bounds__(256) void k_pool(const half_t* __restrict__ g,
                                              float* __restrict__ p) {
    __shared__ float red[256];
    const int c = blockIdx.x, b = blockIdx.y, tid = threadIdx.x;
    const half_t* gp = g + ((size_t)b * CC + c) * LL;
    float s = 0.f;
    for (int i = tid; i < LL; i += 256) s += (float)gp[i];
    red[tid] = s;
    __syncthreads();
    for (int off = 128; off > 0; off >>= 1) {
        if (tid < off) red[tid] += red[tid + off];
        __syncthreads();
    }
    if (tid == 0) p[b * CC + c] = red[0] * (1.0f / LL);
}

// ---------------- K3: att = sigmoid(p ws^T + bs); w2s[b] = w2 * diag(att) (f16) ----
// grid = 16, block = 256
__global__ __launch_bounds__(256) void k_att_scale(const float* __restrict__ p,
                                                   const float* __restrict__ wsm,
                                                   const float* __restrict__ bs,
                                                   const float* __restrict__ w2,
                                                   half_t* __restrict__ w2s) {
    __shared__ float pl[256];
    __shared__ float att[256];
    const int b = blockIdx.x, tid = threadIdx.x;
    pl[tid] = p[b * CC + tid];
    __syncthreads();
    float dot = bs[tid];
    const float* wr = wsm + tid * CC;
    for (int c = 0; c < CC; ++c) dot += wr[c] * pl[c];
    att[tid] = sigmoidf_(dot);
    __syncthreads();
    half_t* dst = w2s + (size_t)b * CC * CC;
    for (int idx = tid; idx < CC * CC; idx += 256)
        dst[idx] = (half_t)(w2[idx] * att[idx & (CC - 1)]);
}

// ---------------- K4: pw2 GEMM (WMMA) + bias + residual ----------------
// grid = (64, 16), block = 256 (8 waves)
__global__ __launch_bounds__(256) void k_pw2(const half_t* __restrict__ g,
                                             const half_t* __restrict__ w2s,
                                             const float* __restrict__ b2,
                                             const float* __restrict__ x,
                                             float* __restrict__ out) {
    extern __shared__ char smem[];
    half_t* aLds = (half_t*)smem;                       // [256][264] (m, k)
    half_t* gt   = (half_t*)(smem + CC * A_STRIDE * 2); // [128][264] (n, k) transposed

    const int tid = threadIdx.x;
    const int b   = blockIdx.y;
    const int l0  = blockIdx.x * NT2;

    // stage A = w2s[b] into padded LDS — async DMA path when available
    const half_t* asrc = w2s + (size_t)b * CC * CC;
#ifdef USE_ASYNC_LDS
    {
        half_t* asrc_nc = const_cast<half_t*>(asrc);
        for (int idx = tid; idx < (CC * CC) / 8; idx += 256) {
            int r = idx >> 5;         // row (256 cols = 32 chunks of 8 halfs)
            int c = (idx & 31) * 8;
            __builtin_amdgcn_global_load_async_to_lds_b128(
                (gas_v4i_t)(asrc_nc + r * CC + c),
                (las_v4i_t)(aLds + r * A_STRIDE + c), 0, 0);
        }
    }
#else
    for (int idx = tid; idx < (CC * CC) / 8; idx += 256) {
        int r = idx >> 5;
        int c = (idx & 31) * 8;
        *(v8h*)(aLds + r * A_STRIDE + c) = *(const v8h*)(asrc + r * CC + c);
    }
#endif
    // stage B = g[b][:, l0:l0+128] transposed -> gt[n][k]  (manual: needs transpose)
    for (int idx = tid; idx < CC * (NT2 / 8); idx += 256) {
        int c  = idx >> 4;            // channel (k) — 128/8 = 16 chunks per channel
        int lc = (idx & 15) * 8;
        v8h v = *(const v8h*)(g + ((size_t)b * CC + c) * LL + l0 + lc);
#pragma unroll
        for (int j = 0; j < 8; ++j) gt[(lc + j) * GT_STRIDE + c] = v[j];
    }
#ifdef USE_ASYNC_LDS
    __builtin_amdgcn_s_wait_asynccnt(0);
#endif
    __syncthreads();

    // GEMM: M=256, K=256, N=128; m-tile outer, A-frag reused over 8 n-tiles
    const int wave = tid >> 5;
    const int lane = tid & 31;
    const int ln   = lane & 15;
    const int lh   = lane >> 4;
    for (int mt = wave; mt < 16; mt += 8) {    // 2 m-tiles per wave
        const int m0 = mt * 16;
        v8f acc[8] = {};
        const half_t* arow = aLds + (m0 + ln) * A_STRIDE;
#pragma unroll
        for (int k0 = 0; k0 < CC; k0 += 32) {
            v16h a = load_frag_a(arow + k0, lh);
#pragma unroll
            for (int nt = 0; nt < 8; ++nt) {
                v16h bfr = load_frag_b(gt + (nt * 16 + ln) * GT_STRIDE + k0, lh);
                acc[nt] = wmma_f16(a, bfr, acc[nt]);
            }
        }
        float bias[8];
#pragma unroll
        for (int v = 0; v < 8; ++v) bias[v] = b2[m0 + v + 8 * lh];
#pragma unroll
        for (int nt = 0; nt < 8; ++nt) {
            const int lg = l0 + nt * 16 + ln;
#pragma unroll
            for (int v = 0; v < 8; ++v) {
                int row = m0 + v + 8 * lh;
                size_t off = ((size_t)b * CC + row) * LL + lg;
                out[off] = acc[nt][v] + bias[v] + x[off];
            }
        }
    }
}

// ---------------- launcher ----------------
extern "C" void kernel_launch(void* const* d_in, const int* in_sizes, int n_in,
                              void* d_out, int out_size, void* d_ws, size_t ws_size,
                              hipStream_t stream) {
    const float* x     = (const float*)d_in[0];
    const float* gamma = (const float*)d_in[1];
    const float* beta  = (const float*)d_in[2];
    const float* w1    = (const float*)d_in[3];
    const float* b1    = (const float*)d_in[4];
    const float* wd    = (const float*)d_in[5];
    const float* bd    = (const float*)d_in[6];
    const float* wsm   = (const float*)d_in[7];
    const float* bs    = (const float*)d_in[8];
    const float* w2    = (const float*)d_in[9];
    const float* b2    = (const float*)d_in[10];
    float* out = (float*)d_out;

    // workspace carve-up
    char* wsb = (char*)d_ws;
    const size_t G_BYTES   = (size_t)BB * CC * LL * sizeof(half_t);   // 67,108,864
    const size_t W1H_BYTES = (size_t)CC2 * CC * sizeof(half_t);       // 262,144
    const size_t P_BYTES   = (size_t)BB * CC * sizeof(float);         // 16,384
    half_t* g   = (half_t*)wsb;
    half_t* w1h = (half_t*)(wsb + G_BYTES);
    float*  p   = (float*)(wsb + G_BYTES + W1H_BYTES);
    half_t* w2s = (half_t*)(wsb + G_BYTES + W1H_BYTES + P_BYTES);

    const size_t SM1 = (size_t)NT1 * YN_STRIDE * 2 + (size_t)CC2 * PW_STRIDE * 2 +
                       (4 * NT1 * 2 + 2 * NT1) * sizeof(float);       // ~201 KB
    const size_t SM4 = (size_t)CC * A_STRIDE * 2 + (size_t)NT2 * GT_STRIDE * 2; // ~198 KB

    k_convert_w1<<<dim3((CC2 * CC + 255) / 256), dim3(256), 0, stream>>>(w1, w1h);
    k_ln_pw1_dw_gate<<<dim3(N1TILES, BB), dim3(512), SM1, stream>>>(
        x, gamma, beta, w1h, b1, wd, bd, g);
    k_pool<<<dim3(CC, BB), dim3(256), 0, stream>>>(g, p);
    k_att_scale<<<dim3(BB), dim3(256), 0, stream>>>(p, wsm, bs, w2, w2s);
    k_pw2<<<dim3(LL / NT2, BB), dim3(256), SM4, stream>>>(g, w2s, b2, x, out);
}
